// ARSLM_78864189489617
// MI455X (gfx1250) — compile-verified, matrix operand order checked by reading, not run on hardware
//
#include <hip/hip_runtime.h>
#include <cstdint>

// ---------------------------------------------------------------------------
// Problem constants (from reference)
// ---------------------------------------------------------------------------
#define B_DIM 4
#define T_DIM 1024
#define V_DIM 32000
#define E_DIM 512
#define H_DIM 1024
#define K_CTX 2560   // 2H + E
#define EPS   1e-5f

// CDNA5 async global->LDS staging (ISA §15.18.3, ASYNCcnt). Flip to 0 to fall
// back to synchronous uint4 staging if the assembler rejects the mnemonics.
#define ASYNC_STAGING 1

typedef __attribute__((ext_vector_type(16))) __bf16 bf16x16;
typedef __attribute__((ext_vector_type(8)))  float  f32x8;

static __device__ inline void zero8(f32x8& v) {
#pragma unroll
  for (int j = 0; j < 8; ++j) v[j] = 0.0f;
}

// D = A(16x32 bf16) * B(32x16 bf16) + C(16x16 f32)
static __device__ inline f32x8 wmma_bf16(bf16x16 a, bf16x16 b, f32x8 c) {
  return __builtin_amdgcn_wmma_f32_16x16x32_bf16(false, a, false, b,
                                                 (short)0, c, false, false);
}

// On amdgcn the generic->LDS address conversion is a low-32-bit truncate, so
// the low 32 bits of a generic pointer to __shared__ are the LDS byte offset.
static __device__ inline uint32_t lds_off_u32(const void* p) {
  return (uint32_t)(uintptr_t)p;
}

// Per-lane 16B async copy global->LDS (GLOBAL_LOAD_ASYNC_TO_LDS_B128, GV mode)
static __device__ inline void async_ld_b128(uint32_t lds_off, const void* g) {
#if ASYNC_STAGING
  asm volatile("global_load_async_to_lds_b128 %0, %1, off"
               :: "v"(lds_off), "v"((unsigned long long)(uintptr_t)g)
               : "memory");
#else
  (void)lds_off; (void)g;
#endif
}
static __device__ inline void wait_async0() {
#if ASYNC_STAGING
  asm volatile("s_wait_asynccnt 0x0" ::: "memory");
#endif
}

// ---------------------------------------------------------------------------
// WMMA fragment loaders (CDNA5 16-bit layouts, ISA 7.12.2)
// A 16x32 row-major: lane r=lane&15 holds row r; kh=(lane>>4)*8;
//   elems[0..7]=K kh..kh+7, elems[8..15]=K 16+kh..+7 -> two 16B loads/lane.
// B is PRE-PACKED into fragment-native 512-bf16 records -> two 16B loads/lane.
// C/D 16x16: vgpr j -> row j + (lane>>4)*8, col lane&15.
// ---------------------------------------------------------------------------
static __device__ inline bf16x16 load_frag_A_rm16(const __bf16* p, int ld) {
  const int lane = threadIdx.x & 31;
  const int r  = lane & 15;
  const int kh = (lane >> 4) << 3;
  union { uint4 u[2]; bf16x16 v; } uu;
  uu.u[0] = *(const uint4*)(p + r * ld + kh);
  uu.u[1] = *(const uint4*)(p + r * ld + kh + 16);
  return uu.v;
}

static __device__ inline bf16x16 load_frag_B_pk(const __bf16* frag_base) {
  const int lane = threadIdx.x & 31;
  union { uint4 u[2]; bf16x16 v; } uu;
  const uint4* p = (const uint4*)(frag_base + lane * 16);
  uu.u[0] = p[0];
  uu.u[1] = p[1];
  return uu.v;
}

static __device__ inline float wave_allreduce(float x) {
#pragma unroll
  for (int off = 16; off > 0; off >>= 1) x += __shfl_xor(x, off, 32);
  return x;
}

// ---------------------------------------------------------------------------
// K1: pack fp32 KxN row-major weight -> bf16 WMMA-B fragment records.
// Record f = kt*(N/16)+nt holds 512 bf16 in [lane][elem] register order.
// ---------------------------------------------------------------------------
__global__ void pack_B_kernel(const float* __restrict__ src,
                              __bf16* __restrict__ dst, int K, int N) {
  const int o = blockIdx.x * blockDim.x + threadIdx.x;
  if (o >= K * N) return;
  const int i    = o & 15;
  const int l    = (o >> 4) & 31;
  const int frag = o >> 9;
  const int NT   = N >> 4;
  const int nt = frag % NT, kt = frag / NT;
  const int c = l & 15, kh = (l >> 4) << 3;
  const int k = kt * 32 + ((i < 8) ? (kh + i) : (16 + kh + (i - 8)));
  const int n = nt * 16 + c;
  dst[o] = (__bf16)src[(size_t)k * N + n];
}

// ---------------------------------------------------------------------------
// K2: embedding gather -> bf16, (B,T,E) row-major with row g = b*T + t
// ---------------------------------------------------------------------------
__global__ __launch_bounds__(128) void embed_kernel(const int* __restrict__ x,
                                                    const float* __restrict__ emb,
                                                    __bf16* __restrict__ eb) {
  const int g = blockIdx.x;
  const int tok = x[g];
  const float* s = emb + (size_t)tok * E_DIM;
  __bf16* d = eb + (size_t)g * E_DIM;
  for (int i = threadIdx.x; i < E_DIM; i += 128) d[i] = (__bf16)s[i];
}

// ---------------------------------------------------------------------------
// K3: recurrent scan. One workgroup (32 wave32s), sequential over T.
// Per-step GEMMs use WMMA with fragment-group preloads so B loads issue as
// clauses and later loads hide under earlier WMMAs.
// ---------------------------------------------------------------------------
__global__ __launch_bounds__(1024, 1) void scan_kernel(
    const __bf16* __restrict__ eb,
    const __bf16* __restrict__ W1c_p, const float* __restrict__ b1c,
    const __bf16* __restrict__ W2c_p, const float* __restrict__ b2c,
    const __bf16* __restrict__ Wg1_p, const float* __restrict__ bg1,
    const float*  __restrict__ Wg2,   const float* __restrict__ bg2,
    const __bf16* __restrict__ Wr_p,  const float* __restrict__ br,
    __bf16* __restrict__ states) {
  extern __shared__ char smem_raw[];
  __bf16* ctx16  = (__bf16*)smem_raw;             // 16*2560 (rows 4..15 zero)
  __bf16* relu16 = ctx16 + 16 * K_CTX;            // 16*2048 (rows 4..15 zero)
  __bf16* tanh1  = relu16 + 16 * 2048;            // 4*1024
  float*  cand   = (float*)(tanh1 + 4 * H_DIM);   // 4*1024
  float*  hr     = cand + 4 * H_DIM;              // 4*1024
  float*  red    = hr + 4 * H_DIM;                // 256
  float*  gate   = red + 256;                     // 4
  float*  mv     = gate + 4;                      // 8

  const int tid  = threadIdx.x;
  const int w    = tid >> 5;
  const int lane = tid & 31;

  for (int i = tid; i < 12 * K_CTX; i += 1024) ctx16[4 * K_CTX + i] = (__bf16)0.0f;
  for (int i = tid; i < 12 * 2048; i += 1024) relu16[4 * 2048 + i] = (__bf16)0.0f;

  float hp1[B_DIM], hp2[B_DIM];   // h1 / h2 at column `tid` (registers)
#pragma unroll
  for (int b = 0; b < B_DIM; ++b) { hp1[b] = 0.0f; hp2[b] = 0.0f; }
  __syncthreads();

  for (int t = 0; t < T_DIM; ++t) {
    // ---- refresh ctx rows 0..3 : [h1 | h2 | e_t] as bf16 ----
#pragma unroll
    for (int b = 0; b < B_DIM; ++b) {
      ctx16[b * K_CTX + tid]         = (__bf16)hp1[b];
      ctx16[b * K_CTX + H_DIM + tid] = (__bf16)hp2[b];
    }
    for (int idx = tid; idx < B_DIM * E_DIM; idx += 1024) {
      const int b = idx >> 9, jj = idx & (E_DIM - 1);
      ctx16[b * K_CTX + 2 * H_DIM + jj] =
          eb[(size_t)(b * T_DIM + t) * E_DIM + jj];
    }
    __syncthreads();

    // ---- phase 1: fused ctx@W1c (4 tiles/wave), ctx@Wg1 (2), e@Wr (2) ----
    f32x8 accW[4], accG[2], accR[2];
#pragma unroll
    for (int i = 0; i < 4; ++i) zero8(accW[i]);
#pragma unroll
    for (int i = 0; i < 2; ++i) { zero8(accG[i]); zero8(accR[i]); }

    for (int kt = 0; kt < K_CTX / 32; ++kt) {
      bf16x16 a = load_frag_A_rm16(ctx16 + kt * 32, K_CTX);
      {  // W1c group: 4 frag loads issue as a clause, then 4 WMMAs
        const __bf16* base = W1c_p + (size_t)(kt * 128 + w * 4) * 512;
        bf16x16 bw[4];
#pragma unroll
        for (int i = 0; i < 4; ++i) bw[i] = load_frag_B_pk(base + i * 512);
#pragma unroll
        for (int i = 0; i < 4; ++i) accW[i] = wmma_bf16(a, bw[i], accW[i]);
      }
      {  // Wg1 group
        const __bf16* base = Wg1_p + (size_t)(kt * 64 + w * 2) * 512;
        bf16x16 bg[2];
#pragma unroll
        for (int i = 0; i < 2; ++i) bg[i] = load_frag_B_pk(base + i * 512);
#pragma unroll
        for (int i = 0; i < 2; ++i) accG[i] = wmma_bf16(a, bg[i], accG[i]);
      }
      if (kt >= 64) {  // Wr group (wave-uniform branch)
        const __bf16* base = Wr_p + (size_t)((kt - 64) * 64 + w * 2) * 512;
        bf16x16 brf[2];
#pragma unroll
        for (int i = 0; i < 2; ++i) brf[i] = load_frag_B_pk(base + i * 512);
#pragma unroll
        for (int i = 0; i < 2; ++i) accR[i] = wmma_bf16(a, brf[i], accR[i]);
      }
    }
    // epilogue: batch rows 0..3 live in lanes 0..15, vgprs 0..3
    if (lane < 16) {
#pragma unroll
      for (int i = 0; i < 4; ++i) {
        const int n = (w * 4 + i) * 16 + lane;
        const float bb = b1c[n];
#pragma unroll
        for (int j = 0; j < 4; ++j)
          relu16[j * 2048 + n] = (__bf16)fmaxf(accW[i][j] + bb, 0.0f);
      }
#pragma unroll
      for (int i = 0; i < 2; ++i) {
        const int n = (w * 2 + i) * 16 + lane;
        const float bg = bg1[n];
#pragma unroll
        for (int j = 0; j < 4; ++j) {
          tanh1[j * H_DIM + n] = (__bf16)tanhf(accG[i][j] + bg);
          hr[j * H_DIM + n]    = accR[i][j];
        }
      }
    }
    __syncthreads();

    // ---- phase 2: cand = relu16 @ W2c + b2c  (2 tiles/wave, K=2048) ----
    f32x8 accC[2];
    zero8(accC[0]); zero8(accC[1]);
    for (int kt = 0; kt < 2048 / 32; ++kt) {
      bf16x16 a = load_frag_A_rm16(relu16 + kt * 32, 2048);
      const __bf16* base = W2c_p + (size_t)(kt * 64 + w * 2) * 512;
      bf16x16 bc[2];
#pragma unroll
      for (int i = 0; i < 2; ++i) bc[i] = load_frag_B_pk(base + i * 512);
#pragma unroll
      for (int i = 0; i < 2; ++i) accC[i] = wmma_bf16(a, bc[i], accC[i]);
    }
    if (lane < 16) {
#pragma unroll
      for (int i = 0; i < 2; ++i) {
        const int n = (w * 2 + i) * 16 + lane;
        const float bb = b2c[n];
#pragma unroll
        for (int j = 0; j < 4; ++j) cand[j * H_DIM + n] = accC[i][j] + bb;
      }
    }
    __syncthreads();

    // ---- gate = sigmoid(tanh1 @ Wg2 + bg2) ----
    {
      const int b  = tid >> 8;
      const int l8 = tid & 255;
      float p = 0.0f;
#pragma unroll
      for (int i = 0; i < 4; ++i) {
        const int jj = l8 + 256 * i;
        p += (float)tanh1[b * H_DIM + jj] * Wg2[jj];
      }
      p = wave_allreduce(p);
      if (lane == 0) red[w] = p;
    }
    __syncthreads();
    if (tid < 4) {
      float s = 0.0f;
#pragma unroll
      for (int i = 0; i < 8; ++i) s += red[tid * 8 + i];
      gate[tid] = 1.0f / (1.0f + __expf(-(s + bg2[0])));
    }
    __syncthreads();

    // ---- h update + layernorm (thread owns column tid of all 4 rows) ----
    {
      const int j = tid;
      float hv[B_DIM];
#pragma unroll
      for (int b = 0; b < B_DIM; ++b) {
        hv[b] = hp1[b] + gate[b] * cand[b * H_DIM + j]
              + 0.1f * (hr[b * H_DIM + j] + br[j]);
        float s = wave_allreduce(hv[b]);
        float q = wave_allreduce(hv[b] * hv[b]);
        if (lane == 0) { red[w * 8 + b * 2] = s; red[w * 8 + b * 2 + 1] = q; }
      }
      __syncthreads();
      if (tid < 8) {
        const int b = tid >> 1, which = tid & 1;
        float s = 0.0f;
        for (int ww = 0; ww < 32; ++ww) s += red[ww * 8 + b * 2 + which];
        mv[b * 2 + which] = s;
      }
      __syncthreads();
#pragma unroll
      for (int b = 0; b < B_DIM; ++b) {
        const float mu  = mv[b * 2] * (1.0f / H_DIM);
        const float var = mv[b * 2 + 1] * (1.0f / H_DIM) - mu * mu;
        const float hn  = (hv[b] - mu) * rsqrtf(var + EPS);
        hp2[b] = hp1[b];
        hp1[b] = hn;
        states[(size_t)(b * T_DIM + t) * H_DIM + j] = (__bf16)hn;
      }
      __syncthreads();
    }
  }
}

// ---------------------------------------------------------------------------
// K4/K6: tiled WMMA GEMM  C(MxN,f32) = A(MxK,bf16) @ Bp(packed bf16) + bias
// 256 threads / 8 waves; block tile 64x128, BK=32; wave tile 16x64.
// Double-buffered LDS tiles streamed with global_load_async_to_lds_b128
// (ASYNCcnt) so tile k+1 loads overlap tile k compute.
// ---------------------------------------------------------------------------
__global__ __launch_bounds__(256) void wmma_gemm_bias_kernel(
    const __bf16* __restrict__ A, const __bf16* __restrict__ Bp,
    const float* __restrict__ bias, float* __restrict__ C,
    int M, int N, int K) {
  __shared__ __bf16 As[2][64 * 32];
  __shared__ __bf16 Bs[2][8 * 512];
  const int tid  = threadIdx.x;
  const int wave = tid >> 5, lane = tid & 31;
  const int wm   = wave & 3, wn = wave >> 2;         // 4 x 2 wave grid
  const int m0   = blockIdx.x * 64;
  const int n0   = blockIdx.y * 128;
  const int NT   = N >> 4;
  const int KT   = K >> 5;

  f32x8 acc[4];
#pragma unroll
  for (int i = 0; i < 4; ++i) zero8(acc[i]);

  auto stage = [&](int buf, int kt) {
    const __bf16* ga = A + (size_t)(m0 + (tid >> 2)) * K + kt * 32 + (tid & 3) * 8;
    const __bf16* gb = Bp + (size_t)(kt * NT + (n0 >> 4)) * 512;
#if ASYNC_STAGING
    async_ld_b128(lds_off_u32(&As[buf][0]) + tid * 16, ga);
    async_ld_b128(lds_off_u32(&Bs[buf][0]) + tid * 16, gb + tid * 8);
    async_ld_b128(lds_off_u32(&Bs[buf][0]) + (tid + 256) * 16,
                  gb + (tid + 256) * 8);
#else
    ((uint4*)&As[buf][0])[tid]       = *(const uint4*)ga;
    ((uint4*)&Bs[buf][0])[tid]       = ((const uint4*)gb)[tid];
    ((uint4*)&Bs[buf][0])[tid + 256] = ((const uint4*)gb)[tid + 256];
#endif
  };

  int buf = 0;
  stage(0, 0);
  for (int kt = 0; kt < KT; ++kt) {
    wait_async0();        // tile kt resident in LDS (per-wave ASYNCcnt)
    __syncthreads();      // all waves' copies visible; prev compute done
    if (kt + 1 < KT) stage(buf ^ 1, kt + 1);  // overlap next tile stream-in
    bf16x16 a = load_frag_A_rm16(&As[buf][wm * 16 * 32], 32);
#pragma unroll
    for (int nt = 0; nt < 4; ++nt) {
      bf16x16 b = load_frag_B_pk(&Bs[buf][(wn * 4 + nt) * 512]);
      acc[nt] = wmma_bf16(a, b, acc[nt]);
    }
    buf ^= 1;
  }

  const int c  = lane & 15;
  const int rh = (lane >> 4) * 8;
#pragma unroll
  for (int nt = 0; nt < 4; ++nt) {
    const int n = n0 + wn * 64 + nt * 16 + c;
    const float bb = bias[n];
#pragma unroll
    for (int j = 0; j < 8; ++j) {
      const int gr = m0 + wm * 16 + rh + j;
      C[(size_t)gr * N + n] = acc[nt][j] + bb;
    }
  }
}

// ---------------------------------------------------------------------------
// K5: sliding-window attention (window 9). One wave per (b,t).
// ---------------------------------------------------------------------------
__global__ __launch_bounds__(256) void attn_kernel(const float* __restrict__ q,
                                                   const float* __restrict__ k,
                                                   const float* __restrict__ v,
                                                   __bf16* __restrict__ attn) {
  const int wave = threadIdx.x >> 5, lane = threadIdx.x & 31;
  const int g = blockIdx.x * 8 + wave;
  const int b = g >> 10, t = g & (T_DIM - 1);

  float qv[32];
  const float* qp = q + (size_t)g * H_DIM;
#pragma unroll
  for (int i = 0; i < 32; ++i) qv[i] = qp[lane + 32 * i];

  float sc[9];
#pragma unroll
  for (int w9 = 0; w9 < 9; ++w9) {
    const int idx = t - 8 + w9;            // wave-uniform
    float p = 0.0f;
    if (idx >= 0) {
      const float* kp = k + (size_t)(b * T_DIM + idx) * H_DIM;
#pragma unroll
      for (int i = 0; i < 32; ++i) p += qv[i] * kp[lane + 32 * i];
    }
    p = wave_allreduce(p);
    sc[w9] = (idx >= 0) ? p * 0.03125f : -1e30f;  // 1/sqrt(1024)
  }
  float mx = sc[0];
#pragma unroll
  for (int w9 = 1; w9 < 9; ++w9) mx = fmaxf(mx, sc[w9]);
  float wt[9], den = 0.0f;
#pragma unroll
  for (int w9 = 0; w9 < 9; ++w9) { wt[w9] = __expf(sc[w9] - mx); den += wt[w9]; }
  const float inv = 1.0f / den;

  float o[32];
#pragma unroll
  for (int i = 0; i < 32; ++i) o[i] = 0.0f;
#pragma unroll
  for (int w9 = 0; w9 < 9; ++w9) {
    const int idx = t - 8 + w9;
    if (idx >= 0) {
      const float* vp = v + (size_t)(b * T_DIM + idx) * H_DIM;
      const float ww = wt[w9] * inv;
#pragma unroll
      for (int i = 0; i < 32; ++i) o[i] += ww * vp[lane + 32 * i];
    }
  }
  __bf16* op = attn + (size_t)g * H_DIM;
#pragma unroll
  for (int i = 0; i < 32; ++i) op[lane + 32 * i] = (__bf16)o[i];
}

// ---------------------------------------------------------------------------
// Host orchestration
// ---------------------------------------------------------------------------
extern "C" void kernel_launch(void* const* d_in, const int* in_sizes, int n_in,
                              void* d_out, int out_size, void* d_ws,
                              size_t ws_size, hipStream_t stream) {
  (void)in_sizes; (void)n_in; (void)out_size; (void)ws_size;
  const int*   x    = (const int*)d_in[0];
  const float* emb  = (const float*)d_in[1];
  const float* W1c  = (const float*)d_in[2];
  const float* b1c  = (const float*)d_in[3];
  const float* W2c  = (const float*)d_in[4];
  const float* b2c  = (const float*)d_in[5];
  const float* Wg1  = (const float*)d_in[6];
  const float* bg1  = (const float*)d_in[7];
  const float* Wg2  = (const float*)d_in[8];
  const float* bg2  = (const float*)d_in[9];
  const float* Wr   = (const float*)d_in[10];
  const float* br   = (const float*)d_in[11];
  const float* Wk   = (const float*)d_in[12];
  const float* bk   = (const float*)d_in[13];
  const float* Wq   = (const float*)d_in[14];
  const float* bq   = (const float*)d_in[15];
  const float* Wv   = (const float*)d_in[16];
  const float* bv   = (const float*)d_in[17];
  const float* Wf   = (const float*)d_in[18];
  const float* bf   = (const float*)d_in[19];
  float* out = (float*)d_out;

  char* ws = (char*)d_ws;
  size_t off = 0;
  auto alloc = [&](size_t bytes) -> char* {
    char* p = ws + off;
    off = (off + bytes + 255) & ~(size_t)255;
    return p;
  };
  __bf16* W1c_p  = (__bf16*)alloc((size_t)K_CTX * 2048 * 2);
  __bf16* Wg1_p  = (__bf16*)alloc((size_t)K_CTX * H_DIM * 2);
  __bf16* Wr_p   = (__bf16*)alloc((size_t)E_DIM * H_DIM * 2);
  __bf16* W2c_p  = (__bf16*)alloc((size_t)2048 * H_DIM * 2);
  __bf16* Wq_p   = (__bf16*)alloc((size_t)H_DIM * H_DIM * 2);
  __bf16* Wk_p   = (__bf16*)alloc((size_t)H_DIM * H_DIM * 2);
  __bf16* Wv_p   = (__bf16*)alloc((size_t)H_DIM * H_DIM * 2);
  __bf16* Wf_p   = (__bf16*)alloc((size_t)H_DIM * V_DIM * 2);
  __bf16* e_b    = (__bf16*)alloc((size_t)B_DIM * T_DIM * E_DIM * 2);
  __bf16* st_b   = (__bf16*)alloc((size_t)B_DIM * T_DIM * H_DIM * 2);
  float*  qf     = (float*)alloc((size_t)B_DIM * T_DIM * H_DIM * 4);
  float*  kf     = (float*)alloc((size_t)B_DIM * T_DIM * H_DIM * 4);
  float*  vf     = (float*)alloc((size_t)B_DIM * T_DIM * H_DIM * 4);
  __bf16* attn_b = (__bf16*)alloc((size_t)B_DIM * T_DIM * H_DIM * 2);

  auto pack = [&](const float* s, __bf16* d, int K, int N) {
    pack_B_kernel<<<(K * N + 255) / 256, 256, 0, stream>>>(s, d, K, N);
  };
  pack(W1c, W1c_p, K_CTX, 2048);
  pack(Wg1, Wg1_p, K_CTX, H_DIM);
  pack(Wr,  Wr_p,  E_DIM, H_DIM);
  pack(W2c, W2c_p, 2048,  H_DIM);
  pack(Wq,  Wq_p,  H_DIM, H_DIM);
  pack(Wk,  Wk_p,  H_DIM, H_DIM);
  pack(Wv,  Wv_p,  H_DIM, H_DIM);
  pack(Wf,  Wf_p,  H_DIM, V_DIM);

  embed_kernel<<<B_DIM * T_DIM, 128, 0, stream>>>(x, emb, e_b);

  const size_t scan_smem =
      (size_t)(16 * K_CTX + 16 * 2048 + 4 * H_DIM) * sizeof(__bf16) +
      (size_t)(2 * 4 * H_DIM + 256 + 4 + 8) * sizeof(float);
  scan_kernel<<<1, 1024, scan_smem, stream>>>(e_b, W1c_p, b1c, W2c_p, b2c,
                                              Wg1_p, bg1, Wg2, bg2, Wr_p, br,
                                              st_b);

  const int MR = B_DIM * T_DIM;  // 4096
  wmma_gemm_bias_kernel<<<dim3(MR / 64, H_DIM / 128), 256, 0, stream>>>(
      st_b, Wq_p, bq, qf, MR, H_DIM, H_DIM);
  wmma_gemm_bias_kernel<<<dim3(MR / 64, H_DIM / 128), 256, 0, stream>>>(
      st_b, Wk_p, bk, kf, MR, H_DIM, H_DIM);
  wmma_gemm_bias_kernel<<<dim3(MR / 64, H_DIM / 128), 256, 0, stream>>>(
      st_b, Wv_p, bv, vf, MR, H_DIM, H_DIM);

  attn_kernel<<<MR / 8, 256, 0, stream>>>(qf, kf, vf, attn_b);

  wmma_gemm_bias_kernel<<<dim3(MR / 64, V_DIM / 128), 256, 0, stream>>>(
      attn_b, Wf_p, bf, out, MR, V_DIM, H_DIM);
}
